// FeatAggregate_4303557230725
// MI455X (gfx1250) — matrix operand order repaired and checked by reading, not compiled, not run on hardware
//
#include <hip/hip_runtime.h>
#include <hip/hip_bf16.h>

// Problem sizes (match reference)
#define B_    512
#define T_    128
#define D_    1024
#define H_    128
#define G4    512     // 4*H
#define CHUNK 16      // timesteps per precompute chunk

typedef __attribute__((ext_vector_type(16))) __bf16 v16bf;
typedef __attribute__((ext_vector_type(8)))  float  v8f;

#if defined(__has_builtin)
#if __has_builtin(__builtin_amdgcn_global_load_async_to_lds_b128)
#define HAVE_ASYNC_LDS 1
#endif
#endif

#ifdef HAVE_ASYNC_LDS
// builtin signature (from compiler diagnostic): src is v4i in addrspace(1)
typedef int v4i_vs __attribute__((vector_size(4 * sizeof(int))));
typedef __attribute__((address_space(1))) v4i_vs gl_v4i;   // global
typedef __attribute__((address_space(3))) v4i_vs ld_v4i;   // LDS
#endif

union Frag16 {
    unsigned short s[16];
    unsigned int   u[8];
    v16bf          b;
};

__device__ __forceinline__ unsigned short f32_bf16(float f) {
    unsigned int u = __float_as_uint(f);
    u += 0x7FFFu + ((u >> 16) & 1u);   // round-to-nearest-even
    return (unsigned short)(u >> 16);
}

__device__ __forceinline__ unsigned int f32x2_bf16(float lo, float hi) {
    return (unsigned int)f32_bf16(lo) | ((unsigned int)f32_bf16(hi) << 16);
}

__device__ __forceinline__ float sigmoidf_(float x) {
    return 1.0f / (1.0f + __expf(-x));
}

// A-fragment (16x32 bf16, MxK): lane m=l&15, kh=l>>4.
// elems 0..7  -> K = kb + kh*8 + j ; elems 8..15 -> K = kb + 16 + kh*8 + (j-8)
__device__ __forceinline__ void loadA_bf16(Frag16& f, const unsigned short* row,
                                           int kb, int kh) {
    const uint4* p0 = (const uint4*)(row + kb + kh * 8);
    const uint4* p1 = (const uint4*)(row + kb + 16 + kh * 8);
    uint4 a = *p0, b = *p1;
    f.u[0] = a.x; f.u[1] = a.y; f.u[2] = a.z; f.u[3] = a.w;
    f.u[4] = b.x; f.u[5] = b.y; f.u[6] = b.z; f.u[7] = b.w;
}

// B-fragment (32x16 bf16, KxN): lane n=l&15 fixed column, kh=l>>4.
// elems 0..7 -> K = kb + kh*8 + j ; elems 8..15 -> K = kb + 16 + kh*8 + (j-8) ; row points at W[g][*] (k contiguous)
__device__ __forceinline__ void loadB_bf16(Frag16& f, const unsigned short* row,
                                           int kb, int kh) {
    const uint4* p0 = (const uint4*)(row + kb + kh * 8);
    const uint4* p1 = (const uint4*)(row + kb + 16 + kh * 8);
    uint4 a = *p0, b = *p1;
    f.u[0] = a.x; f.u[1] = a.y; f.u[2] = a.z; f.u[3] = a.w;
    f.u[4] = b.x; f.u[5] = b.y; f.u[6] = b.z; f.u[7] = b.w;
}

// ---------------- small utility kernels ----------------

__global__ void cvt_f32_to_bf16_kernel(const float* __restrict__ src,
                                       unsigned short* __restrict__ dst, int n) {
    int i = blockIdx.x * blockDim.x + threadIdx.x;
    if (i < n) dst[i] = f32_bf16(src[i]);
}

__global__ void bias_sum_kernel(const float* __restrict__ a,
                                const float* __restrict__ b,
                                float* __restrict__ o, int n) {
    int i = blockIdx.x * blockDim.x + threadIdx.x;
    if (i < n) o[i] = a[i] + b[i];
}

__global__ void zero_state_kernel(unsigned short* __restrict__ h1b,
                                  unsigned short* __restrict__ h2b,
                                  float* __restrict__ c1,
                                  float* __restrict__ c2) {
    int i = blockIdx.x * blockDim.x + threadIdx.x;   // n = B*H = 65536
    if (i < B_ * H_) { h1b[i] = 0; h2b[i] = 0; c1[i] = 0.0f; c2[i] = 0.0f; }
}

// Convert one time-chunk of feats to bf16 (pure bandwidth pass; packed stores).
__global__ void __launch_bounds__(256)
cvt_feats_chunk_kernel(const float* __restrict__ feats,
                       unsigned short* __restrict__ featsb, int t0) {
    size_t i = ((size_t)blockIdx.x * blockDim.x + threadIdx.x) * 8;  // 8 elems/thread
    size_t b = i / ((size_t)CHUNK * D_);
    size_t r = i - b * ((size_t)CHUNK * D_);
    const float4* src = (const float4*)(feats + (b * T_ + t0) * (size_t)D_ + r);
    float4 x = src[0], y = src[1];
    uint4 o;
    o.x = f32x2_bf16(x.x, x.y);
    o.y = f32x2_bf16(x.z, x.w);
    o.z = f32x2_bf16(y.x, y.y);
    o.w = f32x2_bf16(y.z, y.w);
    *(uint4*)(featsb + b * (size_t)CHUNK * D_ + r) = o;
}

// ---------------- bulk input GEMM (time-parallel) ----------------
// chunkGates[b][tc][g] = featsb[b][tc][:] . Wih1[g][:] + (bih1[g]+bhh1[g])
__global__ void __launch_bounds__(128)
gates1_gemm_kernel(const unsigned short* __restrict__ featsb,
                   const unsigned short* __restrict__ Wih1b,
                   const float* __restrict__ bias1,
                   float* __restrict__ chunkGates) {
    const int lane = threadIdx.x & 31;
    const int wave = threadIdx.x >> 5;             // 0..3
    const int ln15 = lane & 15;
    const int kh   = lane >> 4;
    const int b    = blockIdx.x;                   // M-tile = 16 timesteps of b
    const int nt0  = blockIdx.y * 16 + wave * 4;   // first of 4 N-tiles

    const unsigned short* A0  = featsb + ((size_t)b * CHUNK + ln15) * D_;
    const unsigned short* Br0 = Wih1b + (size_t)((nt0 + 0) * 16 + ln15) * D_;
    const unsigned short* Br1 = Wih1b + (size_t)((nt0 + 1) * 16 + ln15) * D_;
    const unsigned short* Br2 = Wih1b + (size_t)((nt0 + 2) * 16 + ln15) * D_;
    const unsigned short* Br3 = Wih1b + (size_t)((nt0 + 3) * 16 + ln15) * D_;

    v8f acc0 = {}, acc1 = {}, acc2 = {}, acc3 = {};
    for (int kb = 0; kb < D_; kb += 32) {
        if (kb + 32 < D_) __builtin_prefetch(A0 + kb + 32, 0, 1);
        Frag16 af;
        loadA_bf16(af, A0, kb, kh);
        Frag16 bf0, bf1, bf2, bf3;
        loadB_bf16(bf0, Br0, kb, kh);
        loadB_bf16(bf1, Br1, kb, kh);
        loadB_bf16(bf2, Br2, kb, kh);
        loadB_bf16(bf3, Br3, kb, kh);
        acc0 = __builtin_amdgcn_wmma_f32_16x16x32_bf16(false, af.b, false, bf0.b,
                                                       (short)0, acc0, false, false);
        acc1 = __builtin_amdgcn_wmma_f32_16x16x32_bf16(false, af.b, false, bf1.b,
                                                       (short)0, acc1, false, false);
        acc2 = __builtin_amdgcn_wmma_f32_16x16x32_bf16(false, af.b, false, bf2.b,
                                                       (short)0, acc2, false, false);
        acc3 = __builtin_amdgcn_wmma_f32_16x16x32_bf16(false, af.b, false, bf3.b,
                                                       (short)0, acc3, false, false);
    }
#pragma unroll
    for (int tt = 0; tt < 4; ++tt) {
        v8f acc = (tt == 0) ? acc0 : (tt == 1) ? acc1 : (tt == 2) ? acc2 : acc3;
        const int g = (nt0 + tt) * 16 + ln15;
        const float bias = bias1[g];
        float* out = chunkGates + (size_t)b * CHUNK * G4 + g;
#pragma unroll
        for (int v = 0; v < 8; ++v) {
            int row = kh * 8 + v;                  // tc within chunk
            out[(size_t)row * G4] = acc[v] + bias;
        }
    }
}

// ---------------- persistent per-chunk recurrence kernel ----------------
// One launch runs 16 sequential timesteps. Block owns 16 batch rows end-to-end:
// h1/h2 live in LDS (bf16), c1/c2 in registers, Whh1 & Wih2 staged in LDS.
// LDS: 128K (Whh1) + 128K (Wih2) + 32K (gates) + 4K + 4K = 296 KB of 320 KB.
__global__ void __launch_bounds__(256)
lstm_chunk_kernel(const float* __restrict__ chunkGates,
                  const unsigned short* __restrict__ Whh1b,
                  const unsigned short* __restrict__ Wih2b,
                  const unsigned short* __restrict__ Whh2b,
                  const float* __restrict__ bias2,
                  unsigned short* __restrict__ h1b,
                  unsigned short* __restrict__ h2b,
                  float* __restrict__ c1,
                  float* __restrict__ c2,
                  float* __restrict__ outF, int is_last_chunk) {
    __shared__ unsigned short sWhh1[G4 * H_];   // 128 KB
    __shared__ unsigned short sWih2[G4 * H_];   // 128 KB
    __shared__ float          sG[16][G4];       // 32 KB gate staging
    __shared__ unsigned short sH1[16][H_];      // 4 KB
    __shared__ unsigned short sH2[16][H_];      // 4 KB

    const int tid  = threadIdx.x;
    const int lane = tid & 31;
    const int wave = tid >> 5;        // 0..7
    const int ln15 = lane & 15;
    const int kh   = lane >> 4;
    const int b0   = blockIdx.x * 16;

    // stage recurrent weights into LDS (async global->LDS DMA, ASYNCcnt-tracked)
    {
        const uint4* s1 = (const uint4*)Whh1b;
        const uint4* s2 = (const uint4*)Wih2b;
        uint4* d1 = (uint4*)sWhh1;
        uint4* d2 = (uint4*)sWih2;
#ifdef HAVE_ASYNC_LDS
        for (int i = tid; i < (G4 * H_) / 8; i += 256) {
            __builtin_amdgcn_global_load_async_to_lds_b128(
                (gl_v4i*)(s1 + i), (ld_v4i*)(d1 + i), 0, 0);
            __builtin_amdgcn_global_load_async_to_lds_b128(
                (gl_v4i*)(s2 + i), (ld_v4i*)(d2 + i), 0, 0);
        }
#else
        for (int i = tid; i < (G4 * H_) / 8; i += 256) { d1[i] = s1[i]; d2[i] = s2[i]; }
#endif
    }
    // load h state into LDS, c state into registers (overlaps async weight DMA)
    float c1r[8], c2r[8];
#pragma unroll
    for (int e = 0; e < 8; ++e) {
        int idx = tid * 8 + e;
        int row = idx >> 7;
        int col = idx & (H_ - 1);
        size_t p = (size_t)(b0 + row) * H_ + col;
        sH1[row][col] = h1b[p];
        sH2[row][col] = h2b[p];
        c1r[e] = c1[p];
        c2r[e] = c2[p];
    }
#ifdef HAVE_ASYNC_LDS
#if __has_builtin(__builtin_amdgcn_s_wait_asynccnt)
    __builtin_amdgcn_s_wait_asynccnt(0);
#else
    asm volatile("s_wait_asynccnt 0x0" ::: "memory");
#endif
#endif
    __syncthreads();

#pragma unroll 1
    for (int tc = 0; tc < CHUNK; ++tc) {
        // ----- layer 1: gates1 = pregates + h1 @ Whh1^T -----
#pragma unroll
        for (int tt = 0; tt < 4; ++tt) {
            const int g = (wave * 4 + tt) * 16 + ln15;
            v8f acc;
#pragma unroll
            for (int v = 0; v < 8; ++v) {
                int row = kh * 8 + v;
                acc[v] = chunkGates[((size_t)(b0 + row) * CHUNK + tc) * G4 + g];
            }
            const unsigned short* A0   = &sH1[ln15][0];
            const unsigned short* Brow = sWhh1 + (size_t)g * H_;
#pragma unroll
            for (int kb = 0; kb < H_; kb += 32) {
                Frag16 af, bf;
                loadA_bf16(af, A0, kb, kh);
                loadB_bf16(bf, Brow, kb, kh);
                acc = __builtin_amdgcn_wmma_f32_16x16x32_bf16(
                    false, af.b, false, bf.b, (short)0, acc, false, false);
            }
#pragma unroll
            for (int v = 0; v < 8; ++v) sG[kh * 8 + v][g] = acc[v];
        }
        __syncthreads();

        // ----- layer-1 elementwise (c1 in registers) -----
#pragma unroll
        for (int e = 0; e < 8; ++e) {
            int idx = tid * 8 + e;
            int row = idx >> 7;
            int col = idx & (H_ - 1);
            float ig = sG[row][col];
            float fg = sG[row][H_ + col];
            float gg = sG[row][2 * H_ + col];
            float og = sG[row][3 * H_ + col];
            float cn = sigmoidf_(fg) * c1r[e] + sigmoidf_(ig) * tanhf(gg);
            float hn = sigmoidf_(og) * tanhf(cn);
            c1r[e] = cn;
            sH1[row][col] = f32_bf16(hn);
        }
        __syncthreads();

        // ----- layer 2: gates2 = h1_new @ Wih2^T + h2 @ Whh2^T + bias2 -----
#pragma unroll
        for (int tt = 0; tt < 4; ++tt) {
            const int g = (wave * 4 + tt) * 16 + ln15;
            v8f acc = {};
            const unsigned short* A1 = &sH1[ln15][0];
            const unsigned short* A2 = &sH2[ln15][0];
            const unsigned short* B1 = sWih2 + (size_t)g * H_;
            const unsigned short* B2 = Whh2b + (size_t)g * H_;   // global (L2)
#pragma unroll
            for (int kb = 0; kb < H_; kb += 32) {
                Frag16 a1, b1, a2, b2;
                loadA_bf16(a1, A1, kb, kh);
                loadB_bf16(b1, B1, kb, kh);
                acc = __builtin_amdgcn_wmma_f32_16x16x32_bf16(
                    false, a1.b, false, b1.b, (short)0, acc, false, false);
                loadA_bf16(a2, A2, kb, kh);
                loadB_bf16(b2, B2, kb, kh);
                acc = __builtin_amdgcn_wmma_f32_16x16x32_bf16(
                    false, a2.b, false, b2.b, (short)0, acc, false, false);
            }
            const float bias = bias2[g];
#pragma unroll
            for (int v = 0; v < 8; ++v) sG[kh * 8 + v][g] = acc[v] + bias;
        }
        __syncthreads();

        // ----- layer-2 elementwise (c2 in registers) -----
        const int wr = (is_last_chunk && tc == CHUNK - 1) ? 1 : 0;
#pragma unroll
        for (int e = 0; e < 8; ++e) {
            int idx = tid * 8 + e;
            int row = idx >> 7;
            int col = idx & (H_ - 1);
            float ig = sG[row][col];
            float fg = sG[row][H_ + col];
            float gg = sG[row][2 * H_ + col];
            float og = sG[row][3 * H_ + col];
            float cn = sigmoidf_(fg) * c2r[e] + sigmoidf_(ig) * tanhf(gg);
            float hn = sigmoidf_(og) * tanhf(cn);
            c2r[e] = cn;
            sH2[row][col] = f32_bf16(hn);
            if (wr) outF[(size_t)(b0 + row) * H_ + col] = hn;
        }
        __syncthreads();
    }

    // hand state back to global for the next chunk
#pragma unroll
    for (int e = 0; e < 8; ++e) {
        int idx = tid * 8 + e;
        int row = idx >> 7;
        int col = idx & (H_ - 1);
        size_t p = (size_t)(b0 + row) * H_ + col;
        h1b[p] = sH1[row][col];
        h2b[p] = sH2[row][col];
        c1[p] = c1r[e];
        c2[p] = c2r[e];
    }
}

// ---------------- host-side orchestration ----------------
extern "C" void kernel_launch(void* const* d_in, const int* in_sizes, int n_in,
                              void* d_out, int out_size, void* d_ws, size_t ws_size,
                              hipStream_t stream) {
    (void)in_sizes; (void)n_in; (void)out_size; (void)ws_size;
    const float* feats = (const float*)d_in[0];
    const float* Wih1  = (const float*)d_in[1];
    const float* Whh1  = (const float*)d_in[2];
    const float* bih1  = (const float*)d_in[3];
    const float* bhh1  = (const float*)d_in[4];
    const float* Wih2  = (const float*)d_in[5];
    const float* Whh2  = (const float*)d_in[6];
    const float* bih2  = (const float*)d_in[7];
    const float* bhh2  = (const float*)d_in[8];
    float* out = (float*)d_out;

    char* ws = (char*)d_ws;
    size_t off = 0;
    auto carve = [&](size_t bytes) -> void* {
        void* p = ws + off;
        off += (bytes + 255) & ~(size_t)255;
        return p;
    };
    unsigned short* Wih1b = (unsigned short*)carve((size_t)G4 * D_ * 2);
    unsigned short* Whh1b = (unsigned short*)carve((size_t)G4 * H_ * 2);
    unsigned short* Wih2b = (unsigned short*)carve((size_t)G4 * H_ * 2);
    unsigned short* Whh2b = (unsigned short*)carve((size_t)G4 * H_ * 2);
    float* bias1 = (float*)carve(G4 * 4);
    float* bias2 = (float*)carve(G4 * 4);
    unsigned short* h1b = (unsigned short*)carve((size_t)B_ * H_ * 2);
    unsigned short* h2b = (unsigned short*)carve((size_t)B_ * H_ * 2);
    float* c1 = (float*)carve((size_t)B_ * H_ * 4);
    float* c2 = (float*)carve((size_t)B_ * H_ * 4);
    unsigned short* featsb = (unsigned short*)carve((size_t)B_ * CHUNK * D_ * 2); // 16 MB
    float* chunkGates = (float*)carve((size_t)B_ * CHUNK * G4 * 4);               // 16 MB

    // per-call weight conversion + state init
    cvt_f32_to_bf16_kernel<<<(G4 * D_ + 255) / 256, 256, 0, stream>>>(Wih1, Wih1b, G4 * D_);
    cvt_f32_to_bf16_kernel<<<(G4 * H_ + 255) / 256, 256, 0, stream>>>(Whh1, Whh1b, G4 * H_);
    cvt_f32_to_bf16_kernel<<<(G4 * H_ + 255) / 256, 256, 0, stream>>>(Wih2, Wih2b, G4 * H_);
    cvt_f32_to_bf16_kernel<<<(G4 * H_ + 255) / 256, 256, 0, stream>>>(Whh2, Whh2b, G4 * H_);
    bias_sum_kernel<<<(G4 + 255) / 256, 256, 0, stream>>>(bih1, bhh1, bias1, G4);
    bias_sum_kernel<<<(G4 + 255) / 256, 256, 0, stream>>>(bih2, bhh2, bias2, G4);
    zero_state_kernel<<<(B_ * H_ + 255) / 256, 256, 0, stream>>>(h1b, h2b, c1, c2);

    const int cvtBlocks = (B_ * CHUNK * D_) / 8 / 256;   // 8 elems per thread
    for (int chunk = 0; chunk < T_ / CHUNK; ++chunk) {
        const int t0 = chunk * CHUNK;
        cvt_feats_chunk_kernel<<<cvtBlocks, 256, 0, stream>>>(feats, featsb, t0);
        dim3 gg(B_, 2);  // 512 M-tiles x (2 * 4 waves * 4 tiles) N-tiles
        gates1_gemm_kernel<<<gg, 128, 0, stream>>>(featsb, Wih1b, bias1, chunkGates);
        const int is_last = (chunk == T_ / CHUNK - 1) ? 1 : 0;
        lstm_chunk_kernel<<<B_ / 16, 256, 0, stream>>>(
            chunkGates, Whh1b, Wih2b, Whh2b, bias2,
            h1b, h2b, c1, c2, out, is_last);
    }
}
